// SimpleRNN_83511344103558
// MI455X (gfx1250) — compile-verified, matrix operand order checked by reading, not run on hardware
//
#include <hip/hip_runtime.h>
#include <math.h>

// ---------------------------------------------------------------------------
// SimpleRNN forward for MI455X (gfx1250, wave32, WMMA + TDM).
//   B=256, S=512, V=32000, E=H=1024, O=512
// Pipeline:
//   0) fp32 -> bf16 conversion of emb / W_ih / W_hh / W_clf into workspace
//   1) embed+proj GEMM: xp[BS,H] = bf16(emb[x]) @ W_ih^T + b_ih   (WMMA bf16,
//      W_ih K-chunks staged to LDS via TENSOR_LOAD_TO_LDS)
//   2) persistent RNN scan kernel: 512 steps, W_hh slice pinned in LDS via
//      TDM, device-wide sense-reversing barrier between steps     (WMMA bf16)
//   3) classifier GEMM: out = h_final @ W_clf^T + b_clf           (WMMA bf16)
// ---------------------------------------------------------------------------

#define BB 256
#define SS 512
#define VV 32000
#define EE 1024
#define HH 1024
#define OO 512
#define MM (BB * SS)

typedef __attribute__((ext_vector_type(16))) __bf16 v16bf;
typedef __attribute__((ext_vector_type(8)))  float  v8f;
typedef __attribute__((ext_vector_type(4)))  unsigned int v4u;
typedef __attribute__((ext_vector_type(8)))  unsigned int v8u;

__device__ __forceinline__ unsigned short f2bf(float f) {
  unsigned u = __float_as_uint(f);
  u += 0x7FFFu + ((u >> 16) & 1u);          // round-to-nearest-even
  return (unsigned short)(u >> 16);
}
__device__ __forceinline__ float bf2f(unsigned short h) {
  return __uint_as_float(((unsigned)h) << 16);
}

union FragU { v16bf v; uint4 q[2]; };

// Load a 16x32 bf16 WMMA operand fragment from a row-major [rows, ld] matrix.
// Per CDNA5 ISA layout: lane (r,hi) holds row r, K-chunks [k0+8*hi, +8) and
// [k0+16+8*hi, +8) -> two aligned b128 loads per lane.
__device__ __forceinline__ v16bf load_frag(const unsigned short* rowbase, int ld,
                                           int k0, int r, int hi) {
  FragU f;
  const unsigned short* p = rowbase + (size_t)r * ld + k0 + hi * 8;
  f.q[0] = *(const uint4*)(p);
  f.q[1] = *(const uint4*)(p + 16);
  return f.v;
}

__device__ __forceinline__ v8f wmma_bf16(v16bf a, v16bf b, v8f c) {
  return __builtin_amdgcn_wmma_f32_16x16x32_bf16(false, a, false, b,
                                                 (short)0, c, false, false);
}

// ---------------------------------------------------------------------------
// Tensor Data Mover: 2-D tile load (global -> LDS), 2-group descriptor form.
// Descriptor bits per CDNA5 ISA §8.3/8.4. We define the "tensor" as exactly
// the tile window (tensor_dim == tile_dim) with the real row pitch, so no
// OOB case arises. data_size = 2 bytes (bf16). Issue from ONE wave only;
// EXEC is ignored by TDM. Caller must s_wait_tensorcnt + barrier.
// ---------------------------------------------------------------------------
__device__ __forceinline__ void tdm_load_tile_2d(unsigned lds_off_bytes,
                                                 const void* gptr,
                                                 unsigned tile_d0_elems,
                                                 unsigned tile_d1_rows,
                                                 unsigned long long row_pitch_elems) {
  unsigned long long ga = (unsigned long long)gptr;
  v4u g0;
  g0[0] = 1u;                                        // count=1, user mode
  g0[1] = lds_off_bytes;                             // lds_addr
  g0[2] = (unsigned)(ga & 0xffffffffu);              // global_addr[31:0]
  g0[3] = (unsigned)((ga >> 32) & 0x01ffffffu)       // global_addr[56:32]
          | (2u << 30);                              // type = 2 ("image")
  v8u g1;
  g1[0] = (1u << 16);                                // wg_mask=0, data_size=1 (2B)
  g1[1] = (tile_d0_elems & 0xffffu) << 16;           // tensor_dim0[15:0]
  g1[2] = (tile_d0_elems >> 16)                      // tensor_dim0[31:16]
          | ((tile_d1_rows & 0xffffu) << 16);        // tensor_dim1[15:0]
  g1[3] = (tile_d1_rows >> 16)                       // tensor_dim1[31:16]
          | ((tile_d0_elems & 0xffffu) << 16);       // tile_dim0
  g1[4] = (tile_d1_rows & 0xffffu);                  // tile_dim1; tile_dim2=0
  g1[5] = (unsigned)(row_pitch_elems & 0xffffffffu); // dim0_stride[31:0]
  g1[6] = (unsigned)((row_pitch_elems >> 32) & 0xffffu); // dim0_stride[47:32]
  g1[7] = 0u;                                        // dim1_stride (unused, 2D)
  asm volatile("tensor_load_to_lds %0, %1" :: "s"(g0), "s"(g1) : "memory");
}

// ---------------------------------------------------------------------------
// Kernel 0: fp32 -> bf16 conversion (grid-stride)
// ---------------------------------------------------------------------------
__global__ void f32_to_bf16_kernel(const float* __restrict__ in,
                                   unsigned short* __restrict__ out, int n) {
  for (int i = blockIdx.x * blockDim.x + threadIdx.x; i < n;
       i += gridDim.x * blockDim.x)
    out[i] = f2bf(in[i]);
}

// ---------------------------------------------------------------------------
// Kernel 1: embedding gather + input projection GEMM
//   xp[m, n] = sum_k emb[x[m], k] * W_ih[n, k] + b_ih[n]
// Block: 256 thr (8 waves). Block tile 128M x 64N; wave tile 16M x 64N
// (4 accumulators, A fragment reused across the 4 N tiles).
// W_ih K-chunks (64 rows x 256 K = 32 KB) staged in LDS by the TDM.
// ---------------------------------------------------------------------------
__global__ void __launch_bounds__(256)
embed_proj_kernel(const int* __restrict__ xtok,
                  const unsigned short* __restrict__ embB,
                  const unsigned short* __restrict__ WihB,
                  const float* __restrict__ b_ih,
                  unsigned short* __restrict__ xp) {
  __shared__ __align__(128) unsigned short ldsB[64 * 256];   // 32 KB
  const int nblk = blockIdx.x;                // 0..15  -> 64 cols each
  const int mblk = blockIdx.y;                // 0..1023 -> 128 rows each
  const int w = threadIdx.x >> 5;
  const int lane = threadIdx.x & 31;
  const int r = lane & 15, hi = lane >> 4;
  const int n0 = nblk * 64;
  const int m0 = mblk * 128 + w * 16;

  const unsigned ldsB_off = (unsigned)(size_t)(&ldsB[0]);  // low32 = LDS offset

  const int tok = xtok[m0 + r];
  const unsigned short* arow = embB + (size_t)tok * EE;   // gathered A row

  v8f acc[4];
  acc[0] = {}; acc[1] = {}; acc[2] = {}; acc[3] = {};

  for (int kb = 0; kb < EE; kb += 256) {
    __syncthreads();   // previous chunk's consumers done before overwrite
    if (threadIdx.x == 0) {
      // DMA: 64 rows x 256 bf16, row pitch EE elements
      tdm_load_tile_2d(ldsB_off, WihB + (size_t)n0 * EE + kb, 256u, 64u,
                       (unsigned long long)EE);
      __builtin_amdgcn_s_wait_tensorcnt(0);
    }
    __syncthreads();
    for (int k = 0; k < 256; k += 32) {
      v16bf a = load_frag(arow, EE, kb + k, 0, hi);   // per-lane row ptr
      // independent registers for the 4 B fragments -> batched ds waits
      v16bf b0 = load_frag(ldsB + 0 * 16 * 256, 256, k, r, hi);
      v16bf b1 = load_frag(ldsB + 1 * 16 * 256, 256, k, r, hi);
      v16bf b2 = load_frag(ldsB + 2 * 16 * 256, 256, k, r, hi);
      v16bf b3 = load_frag(ldsB + 3 * 16 * 256, 256, k, r, hi);
      acc[0] = wmma_bf16(a, b0, acc[0]);
      acc[1] = wmma_bf16(a, b1, acc[1]);
      acc[2] = wmma_bf16(a, b2, acc[2]);
      acc[3] = wmma_bf16(a, b3, acc[3]);
    }
  }
#pragma unroll
  for (int nt = 0; nt < 4; ++nt) {
    const int n = n0 + nt * 16 + r;
    const float bias = b_ih[n];
#pragma unroll
    for (int j = 0; j < 8; ++j) {
      const int m = m0 + hi * 8 + j;
      xp[(size_t)m * HH + n] = f2bf(acc[nt][j] + bias);
    }
  }
}

// ---------------------------------------------------------------------------
// Kernel 2: persistent sequential scan.
//   h_{s+1} = tanh(xp[:, s, :] + h_s @ W_hh^T + b_hh)
// Grid: 64 blocks (32 N-slices x 2 M-slices), 256 thr. Each block pins its
// W_hh slice (32 rows x 1024 K bf16 = 64 KB) in LDS via one TDM transfer,
// reused for all 512 steps. Wave tile: 16M x 32N (2 accumulators).
// Device-wide sense-reversing barrier between steps.
// ---------------------------------------------------------------------------
__global__ void __launch_bounds__(256)
rnn_scan_kernel(const unsigned short* __restrict__ xp,
                const unsigned short* __restrict__ WhhB,
                const float* __restrict__ b_hh,
                unsigned short* __restrict__ h0,
                unsigned short* __restrict__ h1,
                int* __restrict__ sync_cnt, int* __restrict__ sync_gen) {
  __shared__ __align__(128) unsigned short ldsW[32 * 1024];  // 64 KB
  const int nb = blockIdx.x & 31;   // N slice: 32 cols
  const int mb = blockIdx.x >> 5;   // M slice: 128 rows
  const int w = threadIdx.x >> 5;
  const int lane = threadIdx.x & 31;
  const int r = lane & 15, hi = lane >> 4;
  const int n0 = nb * 32;
  const int m0 = mb * 128 + w * 16;

  // stage W_hh rows [n0, n0+32), all K — one TDM DMA, reused for 512 steps.
  // Rows are contiguous (pitch == HH), i.e. a 32 KB x 2 linear burst.
  if (threadIdx.x == 0) {
    tdm_load_tile_2d((unsigned)(size_t)(&ldsW[0]),
                     WhhB + (size_t)n0 * HH, 1024u, 32u,
                     (unsigned long long)HH);
    __builtin_amdgcn_s_wait_tensorcnt(0);
  }
  __syncthreads();

  const int nA = n0 + r;
  const int nB = n0 + 16 + r;
  const float bhh0 = b_hh[nA];
  const float bhh1 = b_hh[nB];

  for (int s = 0; s < SS; ++s) {
    const unsigned short* hp = (s & 1) ? h1 : h0;
    unsigned short*       hn = (s & 1) ? h0 : h1;
    v8f acc0 = {};
    v8f acc1 = {};
    for (int k = 0; k < HH; k += 32) {
      v16bf a  = load_frag(hp + (size_t)m0 * HH, HH, k, r, hi);
      v16bf b0 = load_frag(ldsW, 1024, k, r, hi);
      v16bf b1 = load_frag(ldsW + 16 * 1024, 1024, k, r, hi);
      acc0 = wmma_bf16(a, b0, acc0);
      acc1 = wmma_bf16(a, b1, acc1);
    }
#pragma unroll
    for (int j = 0; j < 8; ++j) {
      const int m = m0 + hi * 8 + j;
      const size_t xrow = ((size_t)m * SS + s) * HH;
      float v0 = acc0[j] + bf2f(xp[xrow + nA]) + bhh0;
      float v1 = acc1[j] + bf2f(xp[xrow + nB]) + bhh1;
      hn[(size_t)m * HH + nA] = f2bf(tanhf(v0));
      hn[(size_t)m * HH + nB] = f2bf(tanhf(v1));
    }
    // ---- device-wide sense-reversing barrier between timesteps ----
    __syncthreads();
    if (threadIdx.x == 0) {
      __threadfence();
      int g = __hip_atomic_load(sync_gen, __ATOMIC_ACQUIRE,
                                __HIP_MEMORY_SCOPE_AGENT);
      int arrived = __hip_atomic_fetch_add(sync_cnt, 1, __ATOMIC_ACQ_REL,
                                           __HIP_MEMORY_SCOPE_AGENT);
      if (arrived == (int)gridDim.x - 1) {
        __hip_atomic_store(sync_cnt, 0, __ATOMIC_RELAXED,
                           __HIP_MEMORY_SCOPE_AGENT);
        __hip_atomic_fetch_add(sync_gen, 1, __ATOMIC_RELEASE,
                               __HIP_MEMORY_SCOPE_AGENT);
      } else {
        while (__hip_atomic_load(sync_gen, __ATOMIC_ACQUIRE,
                                 __HIP_MEMORY_SCOPE_AGENT) == g)
          __builtin_amdgcn_s_sleep(2);
      }
    }
    __syncthreads();
  }
}

// ---------------------------------------------------------------------------
// Kernel 3: classifier  out[256,512] = h_final @ W_clf^T + b_clf  (fp32 out)
// 512 wave tiles of 16x16; 64 blocks x 8 waves, one tile per wave.
// ---------------------------------------------------------------------------
__global__ void __launch_bounds__(256)
classifier_kernel(const unsigned short* __restrict__ hfin,
                  const unsigned short* __restrict__ WclfB,
                  const float* __restrict__ b_clf,
                  float* __restrict__ out) {
  const int w = threadIdx.x >> 5;
  const int lane = threadIdx.x & 31;
  const int r = lane & 15, hi = lane >> 4;
  const int t = blockIdx.x * 8 + w;   // 0..511
  const int mt = t >> 5, nt = t & 31;
  const int m0 = mt * 16, n0 = nt * 16;

  v8f acc = {};
  for (int k = 0; k < HH; k += 32) {
    v16bf a = load_frag(hfin + (size_t)m0 * HH, HH, k, r, hi);
    v16bf b = load_frag(WclfB + (size_t)n0 * HH, HH, k, r, hi);
    acc = wmma_bf16(a, b, acc);
  }
  const int n = n0 + r;
  const float bias = b_clf[n];
#pragma unroll
  for (int j = 0; j < 8; ++j) {
    const int m = m0 + hi * 8 + j;
    out[(size_t)m * OO + n] = acc[j] + bias;
  }
}

// ---------------------------------------------------------------------------
extern "C" void kernel_launch(void* const* d_in, const int* in_sizes, int n_in,
                              void* d_out, int out_size, void* d_ws,
                              size_t ws_size, hipStream_t stream) {
  (void)in_sizes; (void)n_in; (void)out_size; (void)ws_size;
  const int*   x     = (const int*)d_in[0];
  const float* emb   = (const float*)d_in[1];
  const float* W_ih  = (const float*)d_in[2];
  const float* b_ih  = (const float*)d_in[3];
  const float* W_hh  = (const float*)d_in[4];
  const float* b_hh  = (const float*)d_in[5];
  const float* W_clf = (const float*)d_in[6];
  const float* b_clf = (const float*)d_in[7];
  float* out = (float*)d_out;

  char* ws = (char*)d_ws;
  size_t off = 0;
  auto take = [&](size_t bytes) -> char* {
    char* p = ws + off;
    off += (bytes + 255) & ~(size_t)255;
    return p;
  };
  unsigned short* embB  = (unsigned short*)take((size_t)VV * EE * 2);  // 64 MB
  unsigned short* WihB  = (unsigned short*)take((size_t)HH * EE * 2);
  unsigned short* WhhB  = (unsigned short*)take((size_t)HH * HH * 2);
  unsigned short* WclfB = (unsigned short*)take((size_t)OO * HH * 2);
  unsigned short* xp    = (unsigned short*)take((size_t)MM * HH * 2);  // 256 MB
  unsigned short* h0    = (unsigned short*)take((size_t)BB * HH * 2);
  unsigned short* h1    = (unsigned short*)take((size_t)BB * HH * 2);
  int* syncb            = (int*)take(256);

  // zero h ping-pong buffers (h_0 = 0) and barrier counters; contiguous region
  hipMemsetAsync(h0, 0, (size_t)BB * HH * 2 * 2 + 256, stream);

  f32_to_bf16_kernel<<<2048, 256, 0, stream>>>(emb, embB, VV * EE);
  f32_to_bf16_kernel<<<512, 256, 0, stream>>>(W_ih, WihB, HH * EE);
  f32_to_bf16_kernel<<<512, 256, 0, stream>>>(W_hh, WhhB, HH * HH);
  f32_to_bf16_kernel<<<256, 256, 0, stream>>>(W_clf, WclfB, OO * HH);

  embed_proj_kernel<<<dim3(16, 1024), 256, 0, stream>>>(x, embB, WihB, b_ih, xp);

  rnn_scan_kernel<<<64, 256, 0, stream>>>(xp, WhhB, b_hh, h0, h1,
                                          syncb, syncb + 1);

  // after 512 steps the final hidden state lives in h0 (s=511 writes h0)
  classifier_kernel<<<64, 256, 0, stream>>>(h0, WclfB, b_clf, out);
}
